// BinarizedLinear_57501022158953
// MI455X (gfx1250) — matrix operand order repaired and checked by reading, not compiled, not run on hardware
//
#include <hip/hip_runtime.h>
#include <stdint.h>

// D = A x B + C WMMA types
typedef __attribute__((ext_vector_type(16))) _Float16 v16h;
typedef __attribute__((ext_vector_type(8)))  float    v8f;
typedef __attribute__((ext_vector_type(4)))  float    v4f;

#define D_DIRS 64
#define OUT_F  2048
#define IN_F   2048
#define TILES_PER_D (OUT_F / 16)       // 128 output tiles per direction
#define WAVES_PER_BLOCK 8
#define BLOCK_THREADS (WAVES_PER_BLOCK * 32)

// act[d, o] = sum_i W[d,o,i] * x[d,i];  out = act > bias ? 1 : 0
// One wave32 computes 16 output features of one direction via
// V_WMMA_F32_16X16X32_F16, streaming the 16x2048 fp32 weight slab once.
__global__ __launch_bounds__(BLOCK_THREADS)
void BinarizedLinear_wmma_kernel(const float* __restrict__ W,
                                 const uint8_t* __restrict__ X,
                                 const float* __restrict__ bias,
                                 float* __restrict__ out)
{
    __shared__ __align__(32) _Float16 xsh[IN_F];   // 4 KiB: x[d,:] as f16 0/1

    const int lane = threadIdx.x & 31;
    const int wave = threadIdx.x >> 5;

    // 8 consecutive tiles per block; 128 tiles per direction => all waves
    // in a block share the same direction d.
    const int block_tile0 = blockIdx.x * WAVES_PER_BLOCK;
    const int d      = block_tile0 / TILES_PER_D;
    const int o_base = ((block_tile0 % TILES_PER_D) + wave) * 16;

    // Stage x[d,:] into LDS as f16 (exact for 0/1).
    const uint8_t* xrow = X + (size_t)d * IN_F;
    for (int i = threadIdx.x; i < IN_F; i += BLOCK_THREADS)
        xsh[i] = (_Float16)(float)xrow[i];
    __syncthreads();

    // A-matrix (16x32 f16) lane layout:
    //   lanes 0-15 : row M=lane,    K = 0..15
    //   lanes 16-31: row M=lane-16, K = 16..31
    const int m  = lane & 15;
    const int kA = (lane < 16) ? 0 : 16;
    // B-matrix (32x16 f16) lane layout: lanes 0-15 K=0..15, lanes 16-31 K=16..31
    const int kB = (lane < 16) ? 0 : 16;

    const float* wrow = W + (size_t)d * OUT_F * IN_F + (size_t)(o_base + m) * IN_F;

    v8f acc = {};

    #pragma unroll 2
    for (int kb = 0; kb < IN_F; kb += 32) {
        // 16 weights for this lane's A fragment: streamed, non-temporal.
        v4f w0 = __builtin_nontemporal_load((const v4f*)(wrow + kb + kA));
        v4f w1 = __builtin_nontemporal_load((const v4f*)(wrow + kb + kA + 4));
        v4f w2 = __builtin_nontemporal_load((const v4f*)(wrow + kb + kA + 8));
        v4f w3 = __builtin_nontemporal_load((const v4f*)(wrow + kb + kA + 12));

        v16h a;
        #pragma unroll
        for (int j = 0; j < 4; ++j) {
            a[j]      = (_Float16)w0[j];
            a[4 + j]  = (_Float16)w1[j];
            a[8 + j]  = (_Float16)w2[j];
            a[12 + j] = (_Float16)w3[j];
        }

        // B fragment: x replicated across all 16 columns -> every column of D
        // equals the matvec result. 16 consecutive halves per lane from LDS.
        v16h b = *(const v16h*)(&xsh[kb + kB]);

        // (neg_a, A, neg_b, B, c_mod, C, reuse_a, reuse_b)
        acc = __builtin_amdgcn_wmma_f32_16x16x32_f16(
                  false, a, false, b, (short)0, acc, false, false);
    }

    // C/D layout: lane L holds column N=L%16; VGPR r holds row M=r (lanes 0-15)
    // or M=r+8 (lanes 16-31). D[m][n] == act[o_base+m] for every n, so lanes 0
    // and 16 together hold all 16 results in their accumulator registers.
    if (lane == 0 || lane == 16) {
        const int mbase = (lane == 0) ? 0 : 8;
        #pragma unroll
        for (int r = 0; r < 8; ++r) {
            const int    o   = o_base + mbase + r;
            const size_t idx = (size_t)d * OUT_F + o;
            out[idx] = (acc[r] > bias[idx]) ? 1.0f : 0.0f;
        }
    }
}

extern "C" void kernel_launch(void* const* d_in, const int* in_sizes, int n_in,
                              void* d_out, int out_size, void* d_ws, size_t ws_size,
                              hipStream_t stream) {
    // setup_inputs order: weight_noise (f32, D*OUT*IN), x (bool/u8, D*IN),
    // bias_noise (f32, D*OUT). Output: bool (D*OUT) -> float 0/1.
    const float*   W    = (const float*)d_in[0];
    const uint8_t* X    = (const uint8_t*)d_in[1];
    const float*   bias = (const float*)d_in[2];
    float*         out  = (float*)d_out;

    const int total_tiles = D_DIRS * TILES_PER_D;          // 8192 waves
    dim3 grid(total_tiles / WAVES_PER_BLOCK);              // 1024 blocks
    dim3 block(BLOCK_THREADS);                             // 256 threads / 8 waves

    BinarizedLinear_wmma_kernel<<<grid, block, 0, stream>>>(W, X, bias, out);
}
